// MoireAttention_37125697306820
// MI455X (gfx1250) — matrix operand order repaired and checked by reading, not compiled, not run on hardware
//
#include <hip/hip_runtime.h>
#include <hip/hip_bf16.h>
#include <cstdint>
#include <cstddef>

// ---------------- problem constants ----------------
#define B_     2
#define N_     2048
#define H_     8
#define DH_    32
#define E_     65536
#define DIN_   256
#define O_     256          // H*DH
#define TO_    768          // 3*O
#define INV_SCALE 0.17677669529663687f    // 1/sqrt(32)
#define LOG_EPS   -13.815510557964274f    // ln(1e-6)
#define NEG_BIG   -1.0e6f

typedef __attribute__((ext_vector_type(16))) __bf16 v16bf;
typedef __attribute__((ext_vector_type(8)))  __bf16 v8bf;
typedef __attribute__((ext_vector_type(8)))  float  v8f;

__device__ __forceinline__ v16bf cat16(v8bf lo, v8bf hi) {
  return __builtin_shufflevector(lo, hi, 0,1,2,3,4,5,6,7,8,9,10,11,12,13,14,15);
}
__device__ __forceinline__ v8f wmma_bf16(v16bf a, v16bf b, v8f c) {
  // D = A(16x32 bf16) * B(32x16 bf16) + C(16x16 f32)
  return __builtin_amdgcn_wmma_f32_16x16x32_bf16(false, a, false, b, (short)0, c, false, false);
}

// ---------------- prep: fp32 -> bf16 ----------------
__global__ void k_cvt_x(const float* __restrict__ x, __bf16* __restrict__ xb, int n) {
  int i = blockIdx.x * blockDim.x + threadIdx.x;
  if (i < n) xb[i] = (__bf16)x[i];
}
// transpose qkv_w (DIN x TO) -> wt (TO x DIN) bf16 so WMMA B-operand loads are contiguous
__global__ void k_cvt_w(const float* __restrict__ w, __bf16* __restrict__ wt) {
  int i = blockIdx.x * blockDim.x + threadIdx.x;
  if (i < TO_ * DIN_) {
    int o = i / DIN_, k = i % DIN_;
    wt[i] = (__bf16)w[k * TO_ + o];
  }
}

// ---------------- QKV projection GEMM (bf16 WMMA, f32 accum) ----------------
// tiles: (B*N/16)=256 row-tiles x (768/16)=48 col-tiles ; one wave per 16x16 tile
__global__ void __launch_bounds__(256)
k_qkv(const __bf16* __restrict__ xb, const __bf16* __restrict__ wt,
      const float* __restrict__ bias,
      __bf16* __restrict__ Qb, __bf16* __restrict__ Kb, __bf16* __restrict__ Vt)
{
  const int lane = threadIdx.x & 31;
  const int wid  = threadIdx.x >> 5;
  const int tile = blockIdx.x * 8 + wid;      // 12288 tiles total
  const int mt = tile / 48;
  const int ot = tile % 48;
  const int lc = lane & 15;
  const bool hiL = lane >= 16;

  const __bf16* arow = xb + (size_t)(mt * 16 + lc) * DIN_ + (hiL ? 8 : 0);
  const int col = ot * 16 + lc;
  const __bf16* brow = wt + (size_t)col * DIN_ + (hiL ? 8 : 0);

  v8f c = {};
  #pragma unroll
  for (int s = 0; s < 8; ++s) {
    const int kb = s * 32;
    v8bf a0 = *(const v8bf*)(arow + kb);
    v8bf a1 = *(const v8bf*)(arow + kb + 16);
    v8bf b0 = *(const v8bf*)(brow + kb);
    v8bf b1 = *(const v8bf*)(brow + kb + 16);
    c = wmma_bf16(cat16(a0, a1), cat16(b0, b1), c);
  }

  const float bv = bias[col];
  const int sss = col >> 8, rem = col & 255;
  const int h = rem >> 5, d = rem & 31;
  #pragma unroll
  for (int r = 0; r < 8; ++r) {
    const int m  = r + (hiL ? 8 : 0);
    const int ng = mt * 16 + m;
    const int bb = ng >> 11, n = ng & (N_ - 1);
    const __bf16 v = (__bf16)(c[r] + bv);
    if (sss == 0)      Qb[(((size_t)(bb * H_ + h)) * N_ + n) * DH_ + d] = v;   // Q row-major (n,d)
    else if (sss == 1) Kb[(((size_t)(bb * H_ + h)) * N_ + n) * DH_ + d] = v;   // K row-major (n,d)
    else               Vt[(((size_t)(bb * H_ + h)) * DH_ + d) * N_ + n] = v;   // V transposed (d,n)
  }
}

// ---------------- edge FFN: ea = ffn(ffn(edge_attr)) ----------------
__global__ void __launch_bounds__(256)
k_edge_ffn(const float* __restrict__ ein,
           const float* __restrict__ w1, const float* __restrict__ b1,
           const float* __restrict__ w2, const float* __restrict__ b2,
           float* __restrict__ eout)
{
  __shared__ float sw1[64], sw2[64], sb1[8], sb2[8];
  if (threadIdx.x < 64) { sw1[threadIdx.x] = w1[threadIdx.x]; sw2[threadIdx.x] = w2[threadIdx.x]; }
  if (threadIdx.x < 8)  { sb1[threadIdx.x] = b1[threadIdx.x]; sb2[threadIdx.x] = b2[threadIdx.x]; }
  __syncthreads();
  const int t = blockIdx.x * blockDim.x + threadIdx.x;
  if (t >= B_ * E_) return;
  float v0[8], v1[8];
  #pragma unroll
  for (int i = 0; i < 8; ++i) v0[i] = ein[(size_t)t * H_ + i];
  #pragma unroll
  for (int pass = 0; pass < 2; ++pass) {
    #pragma unroll
    for (int j = 0; j < 8; ++j) {
      float acc = sb1[j];
      #pragma unroll
      for (int i = 0; i < 8; ++i) acc = fmaf(v0[i], sw1[i * 8 + j], acc);
      v1[j] = fmaxf(acc, 0.f);
    }
    #pragma unroll
    for (int j = 0; j < 8; ++j) {
      float acc = sb2[j];
      #pragma unroll
      for (int i = 0; i < 8; ++i) acc = fmaf(v1[i], sw2[i * 8 + j], acc);
      v0[j] = acc;
    }
  }
  #pragma unroll
  for (int i = 0; i < 8; ++i) eout[(size_t)t * H_ + i] = v0[i];
}

// ---------------- CSR build for edge scatter (counting sort by row u) ----------------
__global__ void k_zero_i32(int* __restrict__ p, int n) {
  int i = blockIdx.x * blockDim.x + threadIdx.x;
  if (i < n) p[i] = 0;
}
__global__ void k_count(const int* __restrict__ ei, int* __restrict__ cnt) {
  int t = blockIdx.x * blockDim.x + threadIdx.x;
  if (t >= B_ * E_) return;
  int b = t / E_, e = t % E_;
  int u = ei[(size_t)b * 2 * E_ + e];
  atomicAdd(&cnt[b * N_ + u], 1);
}
__global__ void k_scan(const int* __restrict__ cnt, int* __restrict__ rowptr, int* __restrict__ fo) {
  if (threadIdx.x != 0) return;
  const int b = blockIdx.x;
  int run = 0;
  for (int i = 0; i < N_; ++i) {
    rowptr[b * (N_ + 1) + i] = run;
    fo[b * N_ + i] = run;
    run += cnt[b * N_ + i];
  }
  rowptr[b * (N_ + 1) + N_] = run;
}
__global__ void k_fill(const int* __restrict__ ei, int* __restrict__ fo,
                       int* __restrict__ sv, int* __restrict__ se) {
  int t = blockIdx.x * blockDim.x + threadIdx.x;
  if (t >= B_ * E_) return;
  int b = t / E_, e = t % E_;
  int u = ei[(size_t)b * 2 * E_ + e];
  int v = ei[(size_t)b * 2 * E_ + E_ + e];
  int pos = atomicAdd(&fo[b * N_ + u], 1);
  sv[(size_t)b * E_ + pos] = v;
  se[(size_t)b * E_ + pos] = e;
}

// ---------------- fused attention: one block = (b, h, 16 query rows) ----------------
// LDS: sS 16x2048 f32 (128KB) + sP 16x2048 bf16 (64KB) + sO 16x32 f32 (2KB) = 194KB/WGP
__global__ void __launch_bounds__(256)
k_attn(const __bf16* __restrict__ Qb, const __bf16* __restrict__ Kb,
       const __bf16* __restrict__ Vt,
       const float* __restrict__ adj, const float* __restrict__ ea,
       const int* __restrict__ rowptr, const int* __restrict__ sv, const int* __restrict__ se,
       const float* __restrict__ shifts, const float* __restrict__ widths,
       const float* __restrict__ selfW, const unsigned char* __restrict__ mask,
       float* __restrict__ out)
{
  extern __shared__ char smem[];
  float*  sS = (float*)smem;                                   // [16][2048] fp32 scores
  __bf16* sP = (__bf16*)(smem + 16 * N_ * 4);                  // [16][2048] bf16 probs
  float*  sO = (float*)(smem + 16 * N_ * 4 + 16 * N_ * 2);     // [16][32]  fp32 out accum

  const int lane = threadIdx.x & 31;
  const int wid  = threadIdx.x >> 5;
  const int qt = blockIdx.x & 127;
  const int bh = blockIdx.x >> 7;
  const int h = bh & 7, b = bh >> 3;
  const int q0 = qt * 16;
  const int lc = lane & 15;
  const bool hiL = lane >= 16;

  sO[threadIdx.x] = 0.f;
  sO[threadIdx.x + 256] = 0.f;

  const float shift  = shifts[h];
  const float wdt    = widths[h];
  const float inv2w2 = 1.f / (2.f * wdt * wdt);
  const float slw    = selfW[h];
  const size_t bhn   = (size_t)(b * H_ + h) * N_;
  const float* adjb  = adj + ((size_t)b * N_) * N_;

  // Q tile (WMMA A operand), loaded once per wave
  const __bf16* qrow = Qb + (bhn + q0 + lc) * DH_ + (hiL ? 8 : 0);
  const v16bf aQ = cat16(*(const v8bf*)qrow, *(const v8bf*)(qrow + 16));

  // ---- phase 1: scores = QK^T/sqrt(d) + log(moire) (+mask), streamed into LDS ----
  for (int kt = wid; kt < 128; kt += 8) {
    const int kcol = kt * 16 + lc;
    const __bf16* krow = Kb + (bhn + kcol) * DH_ + (hiL ? 8 : 0);
    v8f c = {};
    c = wmma_bf16(aQ, cat16(*(const v8bf*)krow, *(const v8bf*)(krow + 16)), c);

    if (kt + 8 < 128)  // prefetch next adj tile (global_prefetch_b8)
      __builtin_prefetch(adjb + (size_t)(q0 + lc) * N_ + (kt + 8) * 16, 0, 1);

    const unsigned char mcol = mask[b * N_ + kcol];
    #pragma unroll
    for (int r = 0; r < 8; ++r) {
      const int ql = r + (hiL ? 8 : 0);
      const int qg = q0 + ql;
      const float a = adjb[(size_t)qg * N_ + kcol];
      const float t = a - shift;
      const float logm = fmaxf(-(t * t) * inv2w2, LOG_EPS);
      float s = c[r] * INV_SCALE + logm;
      if (!(mcol && mask[b * N_ + qg])) s = NEG_BIG;
      sS[ql * N_ + kcol] = s;
    }
  }
  __syncthreads();

  // ---- phase 2: per-wave rows (2 each): edge scatter + diag + softmax + bf16 pack ----
  #pragma unroll
  for (int rr = 0; rr < 2; ++rr) {
    const int ql = wid * 2 + rr;
    const int u  = q0 + ql;
    float* row = sS + ql * N_;

    const int beg = rowptr[b * (N_ + 1) + u];
    const int end = rowptr[b * (N_ + 1) + u + 1];
    for (int j = beg + lane; j < end; j += 32) {       // (u,v) unique per batch -> no conflicts
      const int v = sv[(size_t)b * E_ + j];
      const int e = se[(size_t)b * E_ + j];
      row[v] += ea[((size_t)b * E_ + e) * H_ + h];
    }
    if (lane == 0) row[u] += slw;                      // self-loop diagonal

    float m = -3.0e38f;
    for (int i = lane; i < N_; i += 32) m = fmaxf(m, row[i]);
    #pragma unroll
    for (int o = 16; o > 0; o >>= 1) m = fmaxf(m, __shfl_xor(m, o, 32));
    float sum = 0.f;
    for (int i = lane; i < N_; i += 32) {
      const float p = __expf(row[i] - m);
      row[i] = p;
      sum += p;
    }
    #pragma unroll
    for (int o = 16; o > 0; o >>= 1) sum += __shfl_xor(sum, o, 32);
    const float inv = 1.f / sum;
    __bf16* prow = sP + ql * N_;
    for (int i = lane; i < N_; i += 32) prow[i] = (__bf16)(row[i] * inv);
  }
  __syncthreads();

  // ---- phase 3: out = attn @ V via bf16 WMMA, partials combined with LDS atomics ----
  v8f acc0 = {}, acc1 = {};
  const __bf16* vbase = Vt + (size_t)(b * H_ + h) * DH_ * N_;
  for (int kc = wid; kc < 64; kc += 8) {
    const int koff = kc * 32;
    const __bf16* prow = sP + lc * N_ + koff + (hiL ? 8 : 0);
    const v16bf aP = cat16(*(const v8bf*)prow, *(const v8bf*)(prow + 16)); // ds_load_b128 x2
    const __bf16* v0r = vbase + (size_t)lc * N_        + koff + (hiL ? 8 : 0);
    const __bf16* v1r = vbase + (size_t)(lc + 16) * N_ + koff + (hiL ? 8 : 0);
    acc0 = wmma_bf16(aP, cat16(*(const v8bf*)v0r, *(const v8bf*)(v0r + 16)), acc0);
    acc1 = wmma_bf16(aP, cat16(*(const v8bf*)v1r, *(const v8bf*)(v1r + 16)), acc1);
  }
  #pragma unroll
  for (int r = 0; r < 8; ++r) {
    const int ql = r + (hiL ? 8 : 0);
    atomicAdd(&sO[ql * 32 + lc],      acc0[r]);        // ds_add_f32
    atomicAdd(&sO[ql * 32 + 16 + lc], acc1[r]);
  }
  __syncthreads();

  for (int i = threadIdx.x; i < 512; i += 256) {
    const int ql = i >> 5, d = i & 31;
    out[((size_t)b * N_ + q0 + ql) * O_ + h * DH_ + d] = sO[i];
  }
}

// ---------------- host launcher ----------------
extern "C" void kernel_launch(void* const* d_in, const int* in_sizes, int n_in,
                              void* d_out, int out_size, void* d_ws, size_t ws_size,
                              hipStream_t stream)
{
  const float* x      = (const float*)d_in[0];
  const float* adj    = (const float*)d_in[1];
  const float* eattr  = (const float*)d_in[2];
  const float* qkv_w  = (const float*)d_in[3];
  const float* qkv_b  = (const float*)d_in[4];
  const float* e_w1   = (const float*)d_in[5];
  const float* e_b1   = (const float*)d_in[6];
  const float* e_w2   = (const float*)d_in[7];
  const float* e_b2   = (const float*)d_in[8];
  const float* shifts = (const float*)d_in[9];
  const float* widths = (const float*)d_in[10];
  const float* selfW  = (const float*)d_in[11];
  const int*   ei     = (const int*)d_in[12];
  const unsigned char* mask = (const unsigned char*)d_in[13];
  float* out = (float*)d_out;

  // workspace partition (256B aligned); total ~14 MB
  char* ws = (char*)d_ws;
  size_t off = 0;
  auto take = [&](size_t bytes) -> char* {
    char* p = ws + off;
    off = (off + bytes + 255) & ~(size_t)255;
    return p;
  };
  __bf16* Qb = (__bf16*)take((size_t)B_ * H_ * N_ * DH_ * 2);
  __bf16* Kb = (__bf16*)take((size_t)B_ * H_ * N_ * DH_ * 2);
  __bf16* Vt = (__bf16*)take((size_t)B_ * H_ * N_ * DH_ * 2);
  __bf16* xb = (__bf16*)take((size_t)B_ * N_ * DIN_ * 2);
  __bf16* wt = (__bf16*)take((size_t)TO_ * DIN_ * 2);
  float*  ea = (float*)take((size_t)B_ * E_ * H_ * 4);
  int* cnt    = (int*)take((size_t)B_ * N_ * 4);
  int* rowptr = (int*)take((size_t)B_ * (N_ + 1) * 4);
  int* fo     = (int*)take((size_t)B_ * N_ * 4);
  int* svv    = (int*)take((size_t)B_ * E_ * 4);
  int* see    = (int*)take((size_t)B_ * E_ * 4);
  (void)ws_size; (void)in_sizes; (void)n_in; (void)out_size;

  // prep
  k_cvt_x<<<(B_ * N_ * DIN_ + 255) / 256, 256, 0, stream>>>(x, xb, B_ * N_ * DIN_);
  k_cvt_w<<<(TO_ * DIN_ + 255) / 256, 256, 0, stream>>>(qkv_w, wt);
  // QKV projection: 12288 wave-tiles / 8 waves per block
  k_qkv<<<1536, 256, 0, stream>>>(xb, wt, qkv_b, Qb, Kb, Vt);
  // edge FFN
  k_edge_ffn<<<(B_ * E_ + 255) / 256, 256, 0, stream>>>(eattr, e_w1, e_b1, e_w2, e_b2, ea);
  // CSR build
  k_zero_i32<<<(B_ * N_ + 255) / 256, 256, 0, stream>>>(cnt, B_ * N_);
  k_count<<<(B_ * E_ + 255) / 256, 256, 0, stream>>>(ei, cnt);
  k_scan<<<B_, 32, 0, stream>>>(cnt, rowptr, fo);
  k_fill<<<(B_ * E_ + 255) / 256, 256, 0, stream>>>(ei, fo, svv, see);

  // fused attention: 2048 blocks, 194KB dynamic LDS (CDNA5 WGP has 320KB)
  const size_t smem = (size_t)16 * N_ * 4 + (size_t)16 * N_ * 2 + 16 * 32 * 4;
  (void)hipFuncSetAttribute((const void*)k_attn,
                            hipFuncAttributeMaxDynamicSharedMemorySize, (int)smem);
  k_attn<<<B_ * H_ * (N_ / 16), 256, smem, stream>>>(
      Qb, Kb, Vt, adj, ea, rowptr, svv, see, shifts, widths, selfW, mask, out);
}
